// RNNDecoder_13735305412645
// MI455X (gfx1250) — matrix operand order repaired and checked by reading, not compiled, not run on hardware
//
#include <hip/hip_runtime.h>
#include <hip/hip_bf16.h>
#include <math.h>

typedef _Float16 f16;
typedef __attribute__((ext_vector_type(16))) _Float16 v16h;
typedef __attribute__((ext_vector_type(8)))  float    v8f;

#define NB   2
#define TT   10
#define NN   128
#define NIN_ 6
#define KK   4
#define HH   128
#define EE   (NN*(NN-1))

#define LDA  136   // LDS row stride (halves) for sender tile
#define LDM  136   // LDS row stride (halves) for msg1 tile

__device__ inline v8f vzero8() {
    v8f z;
#pragma unroll
    for (int i = 0; i < 8; i++) z[i] = 0.f;
    return z;
}

__device__ inline float fast_tanh(float x) {
#if __has_builtin(__builtin_amdgcn_tanhf)
    return __builtin_amdgcn_tanhf(x);   // v_tanh_f32 (gfx1250 trans op)
#else
    return tanhf(x);
#endif
}

// ---------------------------------------------------------------------------
// One-time prep:
//  - swizzle message weights into exact B-fragment register order (f16):
//      Wsw[k][kkt][ht][lane][i] = W^T[c = kkt*32 + (i>>3)*16 + (lane>>4)*8 + (i&7)]
//                                   [h = ht*16 + (lane&15)]
//    so a 32x16 fragment is one contiguous, coalesced v16h per lane.
//  - zero hidden state; init rvec[b][n][k][h] = b_msg1[k][h] (hidden0 == 0).
// ---------------------------------------------------------------------------
__global__ void prep_kernel(const float* __restrict__ W_msg1,
                            const float* __restrict__ W_msg2,
                            const float* __restrict__ b_msg1,
                            f16* __restrict__ W1sw, f16* __restrict__ W2sw,
                            float* __restrict__ hidden,
                            float* __restrict__ rvec) {
    int idx = blockIdx.x * 256 + threadIdx.x;
    if (idx < KK * HH * HH) {
        int i    = idx & 15;
        int lane = (idx >> 4) & 31;
        int ht   = (idx >> 9) & 7;
        int kkt  = (idx >> 12) & 3;
        int k    = idx >> 14;
        int c = kkt * 32 + (i >> 3) * 16 + (lane >> 4) * 8 + (i & 7);
        int h = ht * 16 + (lane & 15);
        W1sw[idx] = (f16)W_msg1[((k * HH) + h) * (2 * HH) + HH + c]; // sender half
        W2sw[idx] = (f16)W_msg2[((k * HH) + h) * HH + c];
    }
    if (idx < NB * NN * HH) hidden[idx] = 0.f;
    if (idx < NB * NN * KK * HH) rvec[idx] = b_msg1[idx & (KK * HH - 1)];
}

// ---------------------------------------------------------------------------
// Edge kernel: block = (b, receiver n, half) owning 64 edge rows.
//   msg1 = tanh(h_send @ W1sw[k] + rvec[b,n,k,:])   [64 x 128]  (WMMA)
//   msg2 = tanh(msg1   @ W2sw[k] + b2[k])           [64 x 128]  (WMMA)
//   msum += rel_type[b,e,k]/(K*NIN) * msg2 ; agg_partial = colsum(msum)
// ---------------------------------------------------------------------------
__global__ __launch_bounds__(256) void edge_kernel(
    const float* __restrict__ hidden,
    const float* __restrict__ rel_type,
    const float* __restrict__ rvec,     // [B,N,K,H]
    const f16*  __restrict__ W1sw,      // f16 [K,4,8,32,16] fragment order
    const f16*  __restrict__ W2sw,      // f16 [K,4,8,32,16] fragment order
    const float* __restrict__ b_msg2,   // [K,H]
    float* __restrict__ agg2)           // [2,B,N,H] partial aggregates
{
    __shared__ f16   sA[64 * LDA];      // sender hiddens (f16)
    __shared__ f16   sM[64 * LDM];      // msg1 (f16)
    __shared__ float sRvec[KK * HH];    // receiver contribution + bias, all k
    __shared__ float sRel[64 * 4];      // rel_type/(K*NIN) per local row,k
    __shared__ float sAgg[HH];

    const int tid  = threadIdx.x;
    const int blk  = blockIdx.x;
    const int half = blk & 1;
    const int node = (blk >> 1) & (NN - 1);
    const int b    = blk >> 8;

    if (tid < HH) sAgg[tid] = 0.f;
    for (int idx = tid; idx < KK * HH; idx += 256)
        sRvec[idx] = rvec[((b * NN + node) * KK) * HH + idx];

    // sender tile: local row ml -> global edge row m = half*64+ml; m==127 is a
    // zero-weight pad row (each receiver has only 127 incoming edges).
    for (int idx = tid; idx < 64 * HH; idx += 256) {
        int ml = idx >> 7, h = idx & 127;
        int m = half * 64 + ml;
        f16 v = (f16)0.f;
        if (m < 127) {
            int s = m + (m >= node);
            v = (f16)hidden[(b * NN + s) * HH + h];
        }
        sA[ml * LDA + h] = v;
    }
    for (int idx = tid; idx < 64 * 4; idx += 256) {
        int ml = idx >> 2, k = idx & 3;
        int m = half * 64 + ml;
        float w = 0.f;
        if (m < 127) {
            int s = m + (m >= node);
            int e = s * (NN - 1) + (node < s ? node : node - 1);
            w = rel_type[((b * EE) + e) * KK + k] * (1.0f / (KK * NIN_));
        }
        sRel[ml * 4 + k] = w;
    }
    __syncthreads();

    const int lane  = tid & 31;
    const int wave  = tid >> 5;
    const int strip = (wave & 3) * 16;  // local M base (4 strips of 16 rows)
    const int whalf = wave >> 2;        // column half (0/1) -> 64 cols each
    const int nlane = lane & 15;
    const int mhalf = lane >> 4;

    const v16h* __restrict__ W1v = (const v16h*)W1sw;
    const v16h* __restrict__ W2v = (const v16h*)W2sw;

    v8f msum[4];
#pragma unroll
    for (int t = 0; t < 4; t++) msum[t] = vzero8();

    for (int k = 0; k < KK; k++) {
        // ---- GEMM1: sA(64x128) x W1[k](128x128), wave tile 16x64 ----------
        v8f acc[4];
#pragma unroll
        for (int t = 0; t < 4; t++) acc[t] = vzero8();
#pragma unroll
        for (int kkt = 0; kkt < 4; kkt++) {
            v16h afrag;   // A 16x32 f16 layout (ISA 7.12.2) -> 2x ds_load_b128
#pragma unroll
            for (int i = 0; i < 16; i++) {
                int vg = i >> 1;
                int Kc = ((vg & 4) ? 16 : 0) + (vg & 3) * 2 + (i & 1) + mhalf * 8;
                afrag[i] = sA[(strip + nlane) * LDA + kkt * 32 + Kc];
            }
#pragma unroll
            for (int t = 0; t < 4; t++) {
                v16h bfrag = W1v[(((k * 4) + kkt) * 8 + (whalf * 4 + t)) * 32 + lane];
                acc[t] = __builtin_amdgcn_wmma_f32_16x16x32_f16(
                    false, afrag, false, bfrag, (short)0, acc[t], false, false);
            }
        }
        __syncthreads();   // prior GEMM2 reads of sM retired

        // tanh epilogue -> sM (f16).  C/D layout: M = j + 8*mhalf, N = nlane
#pragma unroll
        for (int t = 0; t < 4; t++) {
            int col = whalf * 64 + t * 16 + nlane;
            float rv = sRvec[k * HH + col];
#pragma unroll
            for (int j = 0; j < 8; j++) {
                int row = strip + j + 8 * mhalf;
                sM[row * LDM + col] = (f16)fast_tanh(acc[t][j] + rv);
            }
        }
        __syncthreads();

        // ---- GEMM2: sM(64x128) x W2[k](128x128) ---------------------------
        v8f acc2[4];
#pragma unroll
        for (int t = 0; t < 4; t++) acc2[t] = vzero8();
#pragma unroll
        for (int kkt = 0; kkt < 4; kkt++) {
            v16h afrag;
#pragma unroll
            for (int i = 0; i < 16; i++) {
                int vg = i >> 1;
                int Kc = ((vg & 4) ? 16 : 0) + (vg & 3) * 2 + (i & 1) + mhalf * 8;
                afrag[i] = sM[(strip + nlane) * LDM + kkt * 32 + Kc];
            }
#pragma unroll
            for (int t = 0; t < 4; t++) {
                v16h bfrag = W2v[(((k * 4) + kkt) * 8 + (whalf * 4 + t)) * 32 + lane];
                acc2[t] = __builtin_amdgcn_wmma_f32_16x16x32_f16(
                    false, afrag, false, bfrag, (short)0, acc2[t], false, false);
            }
        }

        // weighted accumulate: msum += rel * tanh(msg2)
        float wrel[8];
#pragma unroll
        for (int j = 0; j < 8; j++)
            wrel[j] = sRel[(strip + j + 8 * mhalf) * 4 + k];
#pragma unroll
        for (int t = 0; t < 4; t++) {
            int col = whalf * 64 + t * 16 + nlane;
            float b2 = b_msg2[k * HH + col];
#pragma unroll
            for (int j = 0; j < 8; j++)
                msum[t][j] += wrel[j] * fast_tanh(acc2[t][j] + b2);
        }
        __syncthreads();   // protect sM before next k rewrites it
    }

    // column reduction over this block's 64 edge rows
#pragma unroll
    for (int t = 0; t < 4; t++) {
        float p = 0.f;
#pragma unroll
        for (int j = 0; j < 8; j++) p += msum[t][j];
        atomicAdd(&sAgg[whalf * 64 + t * 16 + nlane], p);
    }
    __syncthreads();
    if (tid < HH)
        agg2[((half * NB + b) * NN + node) * HH + tid] = sAgg[tid];
}

// ---------------------------------------------------------------------------
// Node kernel: GRU update + output MLP for one (b, node); 128 threads.
// Also produces rvec[b,n,k,:] = b_msg1[k] + h_new . W_msg1[k][:, :128]
// for the next step's edge kernel.
// ---------------------------------------------------------------------------
__global__ __launch_bounds__(128) void node_kernel(
    const float* __restrict__ data,    // [B,T,N,NIN]
    const float* __restrict__ agg2,    // [2,B,N,H]
    float* __restrict__ hidden,        // [B,N,H] (in/out)
    const float* __restrict__ W_hr, const float* __restrict__ W_hi,
    const float* __restrict__ W_hh,
    const float* __restrict__ W_ir, const float* __restrict__ b_ir,
    const float* __restrict__ W_ii, const float* __restrict__ b_ii,
    const float* __restrict__ W_in, const float* __restrict__ b_in,
    const float* __restrict__ W_o1, const float* __restrict__ b_o1,
    const float* __restrict__ W_o2, const float* __restrict__ b_o2,
    const float* __restrict__ W_o3, const float* __restrict__ b_o3,
    const float* __restrict__ W_msg1,  // [K,H,2H]
    const float* __restrict__ b_msg1,  // [K,H]
    float* __restrict__ rvec,          // [B,N,K,H]
    float* __restrict__ out,           // [B,N,T-1,NIN]
    int t)
{
    __shared__ float sIns[NIN_];
    __shared__ float sAgg[HH];
    __shared__ float sHn[HH];
    __shared__ float sP1[HH];
    __shared__ float sP2[HH];

    int h    = threadIdx.x;
    int node = blockIdx.x & (NN - 1);
    int b    = blockIdx.x >> 7;
    int base = (b * NN + node) * HH;

    if (h < NIN_) sIns[h] = data[((b * TT + t) * NN + node) * NIN_ + h];
    sAgg[h] = agg2[base + h] + agg2[NB * NN * HH + base + h];
    float hOld = hidden[base + h];
    __syncthreads();

    float ar = 0.f, ai = 0.f, ah = 0.f;
    for (int c = 0; c < HH; c++) {
        float a = sAgg[c];
        ar += a * W_hr[h * HH + c];
        ai += a * W_hi[h * HH + c];
        ah += a * W_hh[h * HH + c];
    }
    float xr = b_ir[h], xi = b_ii[h], xn = b_in[h];
    for (int c = 0; c < NIN_; c++) {
        float x = sIns[c];
        xr += x * W_ir[h * NIN_ + c];
        xi += x * W_ii[h * NIN_ + c];
        xn += x * W_in[h * NIN_ + c];
    }
    float r  = 1.f / (1.f + expf(-(xr + ar)));
    float i_ = 1.f / (1.f + expf(-(xi + ai)));
    float n_ = fast_tanh(xn + r * ah);
    float hNew = (1.f - i_) * n_ + i_ * hOld;
    hidden[base + h] = hNew;
    sHn[h] = hNew;
    __syncthreads();

    float p = b_o1[h];
    for (int c = 0; c < HH; c++) p += sHn[c] * W_o1[h * HH + c];
    sP1[h] = fmaxf(p, 0.f);
    __syncthreads();
    p = b_o2[h];
    for (int c = 0; c < HH; c++) p += sP1[c] * W_o2[h * HH + c];
    sP2[h] = fmaxf(p, 0.f);
    __syncthreads();
    if (h < NIN_) {
        float o = sIns[h] + b_o3[h];
        for (int c = 0; c < HH; c++) o += sP2[c] * W_o3[h * HH + c];
        out[((b * NN + node) * (TT - 1) + t) * NIN_ + h] = o;
    }

    // receiver row-vectors for next step's edge kernel
#pragma unroll 1
    for (int k = 0; k < KK; k++) {
        float acc = b_msg1[k * HH + h];
        const float* Wrow = &W_msg1[(k * HH + h) * (2 * HH)];
        for (int c = 0; c < HH; c++) acc += sHn[c] * Wrow[c];
        rvec[((b * NN + node) * KK + k) * HH + h] = acc;
    }
}

// ---------------------------------------------------------------------------
extern "C" void kernel_launch(void* const* d_in, const int* in_sizes, int n_in,
                              void* d_out, int out_size, void* d_ws, size_t ws_size,
                              hipStream_t stream) {
    (void)in_sizes; (void)n_in; (void)out_size; (void)ws_size;
    const float* data     = (const float*)d_in[0];
    const float* rel_type = (const float*)d_in[1];
    // d_in[2]=rel_rec, d_in[3]=rel_send: one-hot — edge structure computed analytically
    const float* W_msg1 = (const float*)d_in[4];
    const float* b_msg1 = (const float*)d_in[5];
    const float* W_msg2 = (const float*)d_in[6];
    const float* b_msg2 = (const float*)d_in[7];
    const float* W_hr = (const float*)d_in[8];
    const float* W_hi = (const float*)d_in[9];
    const float* W_hh = (const float*)d_in[10];
    const float* W_ir = (const float*)d_in[11];
    const float* b_ir = (const float*)d_in[12];
    const float* W_ii = (const float*)d_in[13];
    const float* b_ii = (const float*)d_in[14];
    const float* W_in = (const float*)d_in[15];
    const float* b_in = (const float*)d_in[16];
    const float* W_o1 = (const float*)d_in[17];
    const float* b_o1 = (const float*)d_in[18];
    const float* W_o2 = (const float*)d_in[19];
    const float* b_o2 = (const float*)d_in[20];
    const float* W_o3 = (const float*)d_in[21];
    const float* b_o3 = (const float*)d_in[22];
    float* out = (float*)d_out;

    char* ws = (char*)d_ws;
    f16*   W1sw   = (f16*)(ws);                 // 128 KB
    f16*   W2sw   = (f16*)(ws + 131072);        // 128 KB
    float* hidden = (float*)(ws + 262144);      // 128 KB
    float* agg2   = (float*)(ws + 393216);      // 256 KB
    float* rvec   = (float*)(ws + 655360);      // 512 KB  [B,N,K,H]
    // total workspace: ~1.2 MB

    prep_kernel<<<512, 256, 0, stream>>>(W_msg1, W_msg2, b_msg1,
                                         W1sw, W2sw, hidden, rvec);

    for (int t = 0; t < TT - 1; t++) {
        edge_kernel<<<NB * NN * 2, 256, 0, stream>>>(
            hidden, rel_type, rvec, W1sw, W2sw, b_msg2, agg2);
        node_kernel<<<NB * NN, 128, 0, stream>>>(
            data, agg2, hidden, W_hr, W_hi, W_hh,
            W_ir, b_ir, W_ii, b_ii, W_in, b_in,
            W_o1, b_o1, W_o2, b_o2, W_o3, b_o3,
            W_msg1, b_msg1, rvec, out, t);
    }
}
